// DeFormConvModule_55783035241117
// MI455X (gfx1250) — compile-verified, matrix operand order checked by reading, not run on hardware
//
#include <hip/hip_runtime.h>
#include <hip/hip_bf16.h>

typedef __attribute__((ext_vector_type(16))) __bf16 v16bf;
typedef __attribute__((ext_vector_type(8)))  float  v8f;

#define CIN 256
#define CO  256
#define HDIM 64
#define WDIM 64
#define HW  4096
#define NK2 9
#define KK  2304      // CIN * NK2
#define LDSPITCH 17   // 16 dwords of payload + 1 pad (bank-conflict avoidance)

union Frag { v16bf v; unsigned int u[8]; };

__device__ __forceinline__ unsigned short f2bf(float f) {
    unsigned int u = __float_as_uint(f);
    unsigned int r = u + 0x7FFFu + ((u >> 16) & 1u);   // round-to-nearest-even
    return (unsigned short)(r >> 16);
}

// Pack two fp32 -> packed bf16 pair (low = a, high = b).
// V_PERM_B32 grabs the two high half-words in a single VALU op; the +0x8000
// bias gives round-to-nearest (ties differ from RNE only, immaterial here).
__device__ __forceinline__ unsigned int pack_bf16(float a, float b) {
#if __has_builtin(__builtin_amdgcn_perm)
    unsigned int au = __float_as_uint(a) + 0x8000u;
    unsigned int bu = __float_as_uint(b) + 0x8000u;
    // sel bytes 0..3 pick from src1 (a), 4..7 from src0 (b):
    // result = { b.byte3, b.byte2, a.byte3, a.byte2 }
    return __builtin_amdgcn_perm(bu, au, 0x07060302u);
#else
    return (unsigned int)f2bf(a) | ((unsigned int)f2bf(b) << 16);
#endif
}

// ---------------------------------------------------------------------------
// Kernel 1a: reorder w_dcn [Co][Cin][3][3] fp32 -> bf16 [Co][tap*256 + c]
// ---------------------------------------------------------------------------
__global__ void convert_w_kernel(const float* __restrict__ w_dcn,
                                 unsigned short* __restrict__ wbf) {
    int idx = blockIdx.x * blockDim.x + threadIdx.x;
    if (idx >= CO * KK) return;
    int co   = idx / KK;
    int rem  = idx - co * KK;     // = tap*256 + c
    int ktap = rem >> 8;
    int c    = rem & 255;
    wbf[idx] = f2bf(w_dcn[(co * CIN + c) * NK2 + ktap]);
}

// ---------------------------------------------------------------------------
// Kernel 1b: reorder w_om [27][Cin][3][3] fp32 -> bf16 [32 (pad)][tap*256+c]
// ---------------------------------------------------------------------------
__global__ void convert_wom_kernel(const float* __restrict__ w_om,
                                   unsigned short* __restrict__ womf) {
    int idx = blockIdx.x * blockDim.x + threadIdx.x;
    if (idx >= 32 * KK) return;
    int ch   = idx / KK;
    int rem  = idx - ch * KK;
    int ktap = rem >> 8;
    int c    = rem & 255;
    womf[idx] = (ch < 27) ? f2bf(w_om[(ch * CIN + c) * NK2 + ktap])
                          : (unsigned short)0;
}

// ---------------------------------------------------------------------------
// Kernel 2: offset/mask conv as implicit-GEMM WMMA (M=32pad, K=2304, N=4096/b)
// grid: (32 col tiles, 1, 4 batches), 256 threads = 8 waves.
// Each wave: full M (2x16 subtiles) x 16 columns -> 2 accumulators.
// Validity folded into a multiplicative factor -> unconditional gathers.
// Epilogue fuses py/px/sigmoid(mask) transform.
// ---------------------------------------------------------------------------
__global__ void __launch_bounds__(256)
offset_gemm_kernel(const float* __restrict__ x,
                   const unsigned short* __restrict__ womf,
                   const float* __restrict__ b_om,
                   float* __restrict__ py,
                   float* __restrict__ px,
                   float* __restrict__ msk) {
    __shared__ unsigned int At[32 * LDSPITCH];     // 2.2 KB
    __shared__ unsigned int Bt[128 * LDSPITCH];    // 8.5 KB

    const int tid  = threadIdx.x;
    const int b    = blockIdx.z;
    const int n0   = blockIdx.x * 128;
    const int lane = tid & 31;
    const int wave = tid >> 5;                     // 0..7 -> 16-col slab

    v8f acc[2];
    acc[0] = (v8f){0,0,0,0,0,0,0,0};
    acc[1] = (v8f){0,0,0,0,0,0,0,0};

    // B-tile fill: 2 threads per column, 16 channels each
    const int jcol = tid & 127;
    const int half = tid >> 7;
    const int nn   = n0 + jcol;
    const int ypix = nn >> 6, xpix = nn & 63;
    const float* xb = x + (size_t)b * CIN * HW;

    // A-tile fill: 8 threads per row, 2 dwords each
    const int arow = tid >> 3;                     // 0..31
    const int apos = (tid & 7) * 2;

    for (int ktap = 0; ktap < NK2; ++ktap) {
        // regular-grid im2col position for this tap/column
        int yy  = ypix - 1 + ktap / 3;
        int xx2 = xpix - 1 + ktap % 3;
        bool vld = (yy >= 0) && (yy < HDIM) && (xx2 >= 0) && (xx2 < WDIM);
        float vldf = vld ? 1.0f : 0.0f;            // multiplicative validity
        const float* xcol = xb + (vld ? (yy * WDIM + xx2) : 0);  // safe address

        for (int cc = 0; cc < 8; ++cc) {
            const int kc = ktap * 8 + cc;
            const int c0 = cc * 32;
            // ---- A tile ----
            {
                const uint2* src = (const uint2*)(womf + (size_t)arow * KK + kc * 32);
                uint2 v = src[tid & 7];
                At[arow * LDSPITCH + apos]     = v.x;
                At[arow * LDSPITCH + apos + 1] = v.y;
            }
            // ---- B tile: 16 channels, 1 unconditional load each ----
            {
                const float* xc = xcol + (size_t)(c0 + half * 16) * HW;
                unsigned int* dr = Bt + jcol * LDSPITCH + half * 8;
#pragma unroll
                for (int i = 0; i < 8; ++i) {
                    float v0 = vldf * xc[0]; xc += HW;
                    float v1 = vldf * xc[0]; xc += HW;
                    dr[i] = pack_bf16(v0, v1);
                }
            }
            __syncthreads();
            // ---- fragments (ISA 16-bit layout) + WMMA ----
            const int kb4 = (lane >> 4) * 4;
            Frag afr[2], bfr;
#pragma unroll
            for (int ms = 0; ms < 2; ++ms) {
                const unsigned int* ar = At + (ms * 16 + (lane & 15)) * LDSPITCH;
#pragma unroll
                for (int i = 0; i < 4; ++i) {
                    afr[ms].u[i]     = ar[kb4 + i];
                    afr[ms].u[4 + i] = ar[8 + kb4 + i];
                }
            }
            {
                const unsigned int* br = Bt + (wave * 16 + (lane & 15)) * LDSPITCH;
#pragma unroll
                for (int i = 0; i < 4; ++i) {
                    bfr.u[i]     = br[kb4 + i];
                    bfr.u[4 + i] = br[8 + kb4 + i];
                }
            }
#pragma unroll
            for (int ms = 0; ms < 2; ++ms)
                acc[ms] = __builtin_amdgcn_wmma_f32_16x16x32_bf16(
                    false, afr[ms].v, false, bfr.v, (short)0, acc[ms], false, false);
            __syncthreads();
        }
    }

    // ---- epilogue: bias + py/px/mask transform ----
    const int mhalf = (lane >> 4) * 8;
    const int ncl   = lane & 15;
    const int ncol  = n0 + wave * 16 + ncl;
    const int yo = ncol >> 6, xo = ncol & 63;
#pragma unroll
    for (int ms = 0; ms < 2; ++ms) {
#pragma unroll
        for (int r = 0; r < 8; ++r) {
            int ch = ms * 16 + mhalf + r;
            if (ch >= 27) continue;
            float v = acc[ms][r] + b_om[ch];
            if (ch < 9) {
                py[((b * 9 + ch) * HW) + ncol] = (float)(yo - 1 + ch / 3) + v;
            } else if (ch < 18) {
                int k = ch - 9;
                px[((b * 9 + k) * HW) + ncol] = (float)(xo - 1 + k % 3) + v;
            } else {
                int k = ch - 18;
                msk[((b * 9 + k) * HW) + ncol] = 1.0f / (1.0f + __expf(-v));
            }
        }
    }
}

// ---------------------------------------------------------------------------
// Kernel 3: fused bilinear-sample + bf16 WMMA GEMM + bias + ReLU
// grid: (32 col tiles, 1, 4 batches), 512 threads = 16 waves.
// Full M=256 per block so each sampled B tile is built exactly once.
// Wave (wm 0..3, wn 0..3): 64x32 slab -> 4x2 accumulators.
// ---------------------------------------------------------------------------
__global__ void __launch_bounds__(512)
dcn_gemm_kernel(const float* __restrict__ x,
                const unsigned short* __restrict__ wbf,
                const float* __restrict__ b_dcn,
                const float* __restrict__ py,
                const float* __restrict__ px,
                const float* __restrict__ msk,
                float* __restrict__ out) {
    __shared__ unsigned int Atile[256 * LDSPITCH];   // 17.0 KB
    __shared__ unsigned int Btile[128 * LDSPITCH];   //  8.5 KB

    const int tid  = threadIdx.x;
    const int b    = blockIdx.z;
    const int n0   = blockIdx.x * 128;
    const int lane = tid & 31;
    const int wave = tid >> 5;      // 0..15
    const int wm   = wave >> 2;     // 0..3 : 64-row slab
    const int wn   = wave & 3;      // 0..3 : 32-col slab

    v8f acc[4][2];
#pragma unroll
    for (int i = 0; i < 4; ++i)
#pragma unroll
        for (int j = 0; j < 2; ++j)
            acc[i][j] = (v8f){0.f, 0.f, 0.f, 0.f, 0.f, 0.f, 0.f, 0.f};

    // B-tile fill: 4 threads per column, 8 channels each
    const int jcol    = tid & 127;
    const int quarter = (tid >> 7) & 3;
    // A-tile fill: 2 threads per row, 8 dwords each
    const int arow  = tid >> 1;     // 0..255
    const int apart = tid & 1;

    const float* xb = x + (size_t)b * CIN * HW;

    for (int ktap = 0; ktap < NK2; ++ktap) {
        // --- per-tap bilinear setup (hoisted; validity+mask folded into weights)
        const int po = ((b * 9 + ktap) * HW) + n0 + jcol;
        float pyv = py[po], pxv = px[po], m = msk[po];
        float y0f = floorf(pyv), x0f = floorf(pxv);
        float dy = pyv - y0f, dx = pxv - x0f;
        int y0 = (int)y0f, x0 = (int)x0f, y1 = y0 + 1, x1 = x0 + 1;
        bool vy0 = (y0 >= 0) && (y0 < HDIM), vy1 = (y1 >= 0) && (y1 < HDIM);
        bool vx0 = (x0 >= 0) && (x0 < WDIM), vx1 = (x1 >= 0) && (x1 < WDIM);
        int cy0 = min(max(y0, 0), HDIM - 1), cy1 = min(max(y1, 0), HDIM - 1);
        int cx0 = min(max(x0, 0), WDIM - 1), cx1 = min(max(x1, 0), WDIM - 1);
        int i00 = cy0 * WDIM + cx0, i01 = cy0 * WDIM + cx1;
        int i10 = cy1 * WDIM + cx0, i11 = cy1 * WDIM + cx1;
        float w00 = (vy0 && vx0) ? (1.f - dy) * (1.f - dx) * m : 0.f;
        float w01 = (vy0 && vx1) ? (1.f - dy) * dx * m : 0.f;
        float w10 = (vy1 && vx0) ? dy * (1.f - dx) * m : 0.f;
        float w11 = (vy1 && vx1) ? dy * dx * m : 0.f;

        for (int cc = 0; cc < 8; ++cc) {
            const int kc = ktap * 8 + cc;
            const int c0 = cc * 32;
            // ---- A tile: bf16 weights, all 256 rows ----
            {
                const uint4* src = (const uint4*)(wbf + (size_t)arow * KK + kc * 32);
                uint4 v0 = src[apart * 2 + 0];
                uint4 v1 = src[apart * 2 + 1];
                unsigned int* dr = Atile + arow * LDSPITCH + apart * 8;
                dr[0] = v0.x; dr[1] = v0.y; dr[2] = v0.z; dr[3] = v0.w;
                dr[4] = v1.x; dr[5] = v1.y; dr[6] = v1.z; dr[7] = v1.w;
            }
            // ---- B tile: sample 8 channels, pack bf16 pairs ----
            {
                const float* xc = xb + (size_t)(c0 + quarter * 8) * HW;
                unsigned int* dr = Btile + jcol * LDSPITCH + quarter * 4;
#pragma unroll
                for (int i = 0; i < 4; ++i) {
                    float v0 = w00 * xc[i00] + w01 * xc[i01] + w10 * xc[i10] + w11 * xc[i11];
                    xc += HW;
                    float v1 = w00 * xc[i00] + w01 * xc[i01] + w10 * xc[i10] + w11 * xc[i11];
                    xc += HW;
                    dr[i] = pack_bf16(v0, v1);
                }
            }
            __syncthreads();

            // ---- fragments (ISA 16-bit layout) + 8 WMMA ----
            const int kb4 = (lane >> 4) * 4;
            Frag afr[4], bfr[2];
#pragma unroll
            for (int ms = 0; ms < 4; ++ms) {
                const unsigned int* ar =
                    Atile + (wm * 64 + ms * 16 + (lane & 15)) * LDSPITCH;
#pragma unroll
                for (int i = 0; i < 4; ++i) {
                    afr[ms].u[i]     = ar[kb4 + i];
                    afr[ms].u[4 + i] = ar[8 + kb4 + i];
                }
            }
#pragma unroll
            for (int ns = 0; ns < 2; ++ns) {
                const unsigned int* br =
                    Btile + (wn * 32 + ns * 16 + (lane & 15)) * LDSPITCH;
#pragma unroll
                for (int i = 0; i < 4; ++i) {
                    bfr[ns].u[i]     = br[kb4 + i];
                    bfr[ns].u[4 + i] = br[8 + kb4 + i];
                }
            }
#pragma unroll
            for (int ms = 0; ms < 4; ++ms)
#pragma unroll
                for (int ns = 0; ns < 2; ++ns)
                    acc[ms][ns] = __builtin_amdgcn_wmma_f32_16x16x32_bf16(
                        false, afr[ms].v, false, bfr[ns].v,
                        (short)0, acc[ms][ns], false, false);
            __syncthreads();
        }
    }

    // ---- epilogue: bias + ReLU + store (C layout: VGPR r -> M = r + 8*(lane>=16))
    const int mhalf = (lane >> 4) * 8;
    const int ncl   = lane & 15;
#pragma unroll
    for (int ms = 0; ms < 4; ++ms) {
#pragma unroll
        for (int ns = 0; ns < 2; ++ns) {
            int ncol = n0 + wn * 32 + ns * 16 + ncl;
#pragma unroll
            for (int r = 0; r < 8; ++r) {
                int co = wm * 64 + ms * 16 + mhalf + r;
                float v = acc[ms][ns][r] + b_dcn[co];
                v = v > 0.f ? v : 0.f;
                out[(size_t)(b * CO + co) * HW + ncol] = v;
            }
        }
    }
}

// ---------------------------------------------------------------------------
extern "C" void kernel_launch(void* const* d_in, const int* in_sizes, int n_in,
                              void* d_out, int out_size, void* d_ws, size_t ws_size,
                              hipStream_t stream) {
    const float* x     = (const float*)d_in[0];
    const float* w_om  = (const float*)d_in[1];
    const float* b_om  = (const float*)d_in[2];
    const float* w_dcn = (const float*)d_in[3];
    const float* b_dcn = (const float*)d_in[4];
    float* out = (float*)d_out;

    // workspace: wbf | womf | py | px | mask  (~3.0 MB)
    char* ws = (char*)d_ws;
    unsigned short* wbf  = (unsigned short*)ws;                 // 256*2304*2 = 1179648
    unsigned short* womf = (unsigned short*)(ws + 1179648);     //  32*2304*2 =  147456
    float* py  = (float*)(ws + 1179648 + 147456);               // 4*9*4096 f32 each
    float* px  = py + 4 * 9 * HW;
    float* msk = px + 4 * 9 * HW;

    convert_w_kernel<<<(CO * KK + 255) / 256, 256, 0, stream>>>(w_dcn, wbf);
    convert_wom_kernel<<<(32 * KK + 255) / 256, 256, 0, stream>>>(w_om, womf);

    dim3 g2(32, 1, 4);
    offset_gemm_kernel<<<g2, 256, 0, stream>>>(x, womf, b_om, py, px, msk);

    dim3 g3(32, 1, 4);
    dcn_gemm_kernel<<<g3, 512, 0, stream>>>(x, wbf, b_dcn, py, px, msk, out);
}
